// TransformerBlock_11587821765257
// MI455X (gfx1250) — matrix operand order repaired and checked by reading, not compiled, not run on hardware
//
#include <hip/hip_runtime.h>
#include <cstdint>
#include <cstddef>

#ifndef __has_builtin
#define __has_builtin(x) 0
#endif

typedef _Float16 half8 __attribute__((ext_vector_type(8)));
typedef _Float16 v16h  __attribute__((ext_vector_type(16)));
typedef float    v8f   __attribute__((ext_vector_type(8)));
typedef int      v4i   __attribute__((ext_vector_type(4)));

// ---------------- CDNA5 async global->LDS path (builtin confirmed present) ---
#if defined(__HIP_DEVICE_COMPILE__) && __has_builtin(__builtin_amdgcn_global_load_async_to_lds_b128)
#define ASYNC_COPY 1
typedef __attribute__((address_space(1))) v4i gv4i;   // global (AS1)
typedef __attribute__((address_space(3))) v4i lv4i;   // LDS (AS3)
static __device__ __forceinline__ void async_b128(const void* g, void* l) {
  __builtin_amdgcn_global_load_async_to_lds_b128((gv4i*)g, (lv4i*)l, 0, 0);
}
static __device__ __forceinline__ void async_wait0() {
#if __has_builtin(__builtin_amdgcn_s_wait_asynccnt)
  __builtin_amdgcn_s_wait_asynccnt(0);
#else
  asm volatile("s_wait_asynccnt 0" ::: "memory");
#endif
}
#else
#define ASYNC_COPY 0
#endif

// ---------------- problem constants ----------------
#define NTOK 65536   // B*H*W
#define CCH  256
#define HIDN 1024
#define HH   128
#define WW   128

// =====================================================================
// Weight swizzle: W[K][N] fp32 row-major -> f16 WMMA-B fragment layout.
// Fragment f = kb*(N/16)+nb holds 32 lanes x 16 f16 (contiguous per lane):
//   n = nb*16 + lane%16 ; k = kb*32 + (lane/16)*16 + s   (s = 0..15)
// =====================================================================
__global__ __launch_bounds__(256) void swizzle_w(const float* __restrict__ W,
                                                 _Float16* __restrict__ out,
                                                 int K, int N) {
  int idx = blockIdx.x * 256 + threadIdx.x;     // total K*N
  int s  = idx & 15;
  int l  = (idx >> 4) & 31;
  int f  = idx >> 9;
  int nbCount = N >> 4;
  int kb = f / nbCount;
  int nb = f - kb * nbCount;
  int n  = nb * 16 + (l & 15);
  int k  = kb * 32 + ((l >> 4) * 16) + s;
  out[idx] = (_Float16)W[(size_t)k * N + n];
}

__global__ __launch_bounds__(256) void swizzle_qkv(const float* __restrict__ wq,
                                                   const float* __restrict__ wk,
                                                   const float* __restrict__ wv,
                                                   _Float16* __restrict__ out) {
  const int N = 768;
  int idx = blockIdx.x * 256 + threadIdx.x;
  int s  = idx & 15;
  int l  = (idx >> 4) & 31;
  int f  = idx >> 9;
  int nbCount = N >> 4;
  int kb = f / nbCount;
  int nb = f - kb * nbCount;
  int n  = nb * 16 + (l & 15);
  int k  = kb * 32 + ((l >> 4) * 16) + s;
  int part = n >> 8;
  int nn = n & 255;
  const float* Wp = (part == 0) ? wq : (part == 1 ? wk : wv);
  out[idx] = (_Float16)Wp[(size_t)k * 256 + nn];
}

__global__ __launch_bounds__(256) void bias_cat(const float* __restrict__ bq,
                                                const float* __restrict__ bk,
                                                const float* __restrict__ bv,
                                                float* __restrict__ o) {
  int i = blockIdx.x * 256 + threadIdx.x;       // 0..767
  o[i] = (i < 256) ? bq[i] : (i < 512 ? bk[i - 256] : bv[i - 512]);
}

// =====================================================================
// LN1 + NCHW->NHWC transpose. 64 tokens (consecutive w) per block.
// Writes xc (fp32 channels-last, for residual) and n1 (f16).
// =====================================================================
__global__ __launch_bounds__(256) void ln1_transpose(const float* __restrict__ x,
                                                     const float* __restrict__ g,
                                                     const float* __restrict__ b,
                                                     float* __restrict__ xc,
                                                     _Float16* __restrict__ n1h) {
  __shared__ float T[256][65];          // padded: conflict-free
  __shared__ float mu_s[64], rs_s[64];
  const int t = threadIdx.x;
  const int token0 = blockIdx.x * 64;
  const int bimg = token0 >> 14;        // / 16384
  const int hw0  = token0 & 16383;
  const int wi = t & 63;
  for (int c = t >> 6; c < 256; c += 4)
    T[c][wi] = x[((size_t)(bimg * 256 + c) << 14) + hw0 + wi];
  __syncthreads();
  if (t < 64) {
    float s = 0.f, s2 = 0.f;
    for (int c = 0; c < 256; ++c) { float v = T[c][t]; s += v; s2 += v * v; }
    float mu = s * (1.0f / 256.0f);
    float var = s2 * (1.0f / 256.0f) - mu * mu;
    mu_s[t] = mu;
    rs_s[t] = rsqrtf(var + 1e-5f);
  }
  __syncthreads();
  const int c = t;                      // 256 threads == 256 channels
  const float gc = g[c], bc = b[c];
  for (int i = 0; i < 64; ++i) {
    float v = T[c][i];
    size_t o = (size_t)(token0 + i) * 256 + c;
    xc[o] = v;
    n1h[o] = (_Float16)((v - mu_s[i]) * rs_s[i] * gc + bc);
  }
}

// =====================================================================
// Row LayerNorm (channels-last fp32 in, f16 out). One wave per token.
// =====================================================================
__global__ __launch_bounds__(256) void ln_rows(const float* __restrict__ X,
                                               const float* __restrict__ g,
                                               const float* __restrict__ b,
                                               _Float16* __restrict__ Yh) {
  const int lane = threadIdx.x & 31;
  const int token = blockIdx.x * 8 + (threadIdx.x >> 5);
  const float* xp = X + (size_t)token * 256 + lane * 8;
  float4 v0 = *(const float4*)(xp);
  float4 v1 = *(const float4*)(xp + 4);
  float v[8] = {v0.x, v0.y, v0.z, v0.w, v1.x, v1.y, v1.z, v1.w};
  float s = 0.f, s2 = 0.f;
#pragma unroll
  for (int j = 0; j < 8; ++j) { s += v[j]; s2 += v[j] * v[j]; }
#pragma unroll
  for (int off = 16; off; off >>= 1) { s += __shfl_xor(s, off, 32); s2 += __shfl_xor(s2, off, 32); }
  float mu = s * (1.0f / 256.0f);
  float rs = rsqrtf(s2 * (1.0f / 256.0f) - mu * mu + 1e-5f);
  half8 y;
#pragma unroll
  for (int j = 0; j < 8; ++j) {
    int cc = lane * 8 + j;
    y[j] = (_Float16)((v[j] - mu) * rs * g[cc] + b[cc]);
  }
  *(half8*)(Yh + (size_t)token * 256 + lane * 8) = y;
}

// =====================================================================
// 3x3 neighborhood attention. One wave per token; lane owns 8 channels.
// qkvh layout: [token][768] = q|k|v. Writes x1 = xc + attn (fp32).
// =====================================================================
__global__ __launch_bounds__(256) void attn_kernel(const _Float16* __restrict__ qkvh,
                                                   const float* __restrict__ xc,
                                                   float* __restrict__ x1) {
  const int lane = threadIdx.x & 31;
  const int token = blockIdx.x * 8 + (threadIdx.x >> 5);
  const int h = (token >> 7) & 127;
  const int w = token & 127;
  half8 q8 = *(const half8*)(qkvh + (size_t)token * 768 + lane * 8);
  float qf[8];
#pragma unroll
  for (int j = 0; j < 8; ++j) qf[j] = (float)q8[j];

  float sc[9];
  half8 vn[9];
#pragma unroll
  for (int n = 0; n < 9; ++n) {
    int dy = n / 3 - 1, dx = n % 3 - 1;
    int hh = h + dy, ww = w + dx;
    bool valid = ((unsigned)hh < 128u) && ((unsigned)ww < 128u);  // uniform per wave
    float p = 0.f;
#pragma unroll
    for (int j = 0; j < 8; ++j) vn[n][j] = (_Float16)0.0f;
    if (valid) {
      const _Float16* base = qkvh + (size_t)(token + dy * 128 + dx) * 768;
      half8 k8 = *(const half8*)(base + 256 + lane * 8);
      vn[n]    = *(const half8*)(base + 512 + lane * 8);
#pragma unroll
      for (int j = 0; j < 8; ++j) p += qf[j] * (float)k8[j];
    }
#pragma unroll
    for (int off = 16; off; off >>= 1) p += __shfl_xor(p, off, 32);
    sc[n] = valid ? p * 0.0625f : -1e9f;   // 1/sqrt(256)
  }
  float m = sc[0];
#pragma unroll
  for (int n = 1; n < 9; ++n) m = fmaxf(m, sc[n]);
  float e[9], s = 0.f;
#pragma unroll
  for (int n = 0; n < 9; ++n) { e[n] = __expf(sc[n] - m); s += e[n]; }
  float inv = 1.0f / s;
  float o[8] = {};
#pragma unroll
  for (int n = 0; n < 9; ++n) {
    float wn_ = e[n] * inv;
#pragma unroll
    for (int j = 0; j < 8; ++j) o[j] += wn_ * (float)vn[n][j];
  }
  const float* xp = xc + (size_t)token * 256 + lane * 8;
  float* op = x1 + (size_t)token * 256 + lane * 8;
  float4 r0, r1;
  r0.x = xp[0] + o[0]; r0.y = xp[1] + o[1]; r0.z = xp[2] + o[2]; r0.w = xp[3] + o[3];
  r1.x = xp[4] + o[4]; r1.y = xp[5] + o[5]; r1.z = xp[6] + o[6]; r1.w = xp[7] + o[7];
  *(float4*)(op) = r0;
  *(float4*)(op + 4) = r1;
}

// =====================================================================
// WMMA GEMM: Y[M][NOUT](f16) = A[M][K](f16) x Bsw(+bias, optional GELU)
// Block: 128x128 tile, 256 threads = 8 waves (2 x 4); wave tile 64x32.
// K-step 32 via v_wmma_f32_16x16x32_f16; A double-buffered in LDS via
// async global->LDS DMA. 16B-aligned LDS tile => ds_load_b128.
// =====================================================================
template <int KDIM, bool GELU>
__global__ __launch_bounds__(256) void gemm_f16out(const _Float16* __restrict__ A,
                                                   const _Float16* __restrict__ Bsw,
                                                   const float* __restrict__ bias,
                                                   _Float16* __restrict__ Y,
                                                   int NOUT) {
  constexpr int KT = KDIM / 32;
  __shared__ __align__(16) _Float16 As[2][128][40];  // pitch 40: conflict-free b128
  const int tid = threadIdx.x;
  const int lane = tid & 31;
  const int wave = tid >> 5;
  const int wm = wave & 1;
  const int wn = wave >> 1;
  const int rowBase = blockIdx.x * 128;
  const int nbBase = blockIdx.y * 8 + wn * 2;
  v8f acc[4][2] = {};

  auto stage = [&](int buf, int kt) {
    const int k0 = kt * 32;
#pragma unroll
    for (int i = 0; i < 2; ++i) {
      int cid = tid + i * 256;
      int r = cid >> 2;
      int ch = (cid & 3) << 3;
      const _Float16* src = A + (size_t)(rowBase + r) * KDIM + k0 + ch;
      _Float16* dst = &As[buf][r][ch];
#if ASYNC_COPY
      async_b128(src, dst);
#else
      *(half8*)dst = *(const half8*)src;
#endif
    }
  };

  stage(0, 0);
  for (int kt = 0; kt < KT; ++kt) {
    const int buf = kt & 1;
#if ASYNC_COPY
    async_wait0();
#endif
    __syncthreads();
    if (kt + 1 < KT) stage(buf ^ 1, kt + 1);

    const size_t bbase = ((size_t)kt * (NOUT >> 4) + nbBase) * 512 + lane * 16;
    v16h bf0 = *(const v16h*)(Bsw + bbase);
    v16h bf1 = *(const v16h*)(Bsw + bbase + 512);

    const int arow = wm * 64 + (lane & 15);
    const int hof = (lane >> 4) * 16;
    v16h af[4];
#pragma unroll
    for (int mt = 0; mt < 4; ++mt) {
      half8 a0 = *(const half8*)&As[buf][arow + mt * 16][hof];
      half8 a1 = *(const half8*)&As[buf][arow + mt * 16][hof + 8];
      af[mt] = __builtin_shufflevector(a0, a1, 0, 1, 2, 3, 4, 5, 6, 7,
                                       8, 9, 10, 11, 12, 13, 14, 15);
    }
#pragma unroll
    for (int mt = 0; mt < 4; ++mt) {
      acc[mt][0] = __builtin_amdgcn_wmma_f32_16x16x32_f16(false, af[mt], false, bf0,
                                                          (short)0, acc[mt][0], false, false);
      acc[mt][1] = __builtin_amdgcn_wmma_f32_16x16x32_f16(false, af[mt], false, bf1,
                                                          (short)0, acc[mt][1], false, false);
    }
    __syncthreads();
  }

  const int hof = (lane >> 4) << 3;
#pragma unroll
  for (int mt = 0; mt < 4; ++mt) {
#pragma unroll
    for (int nt = 0; nt < 2; ++nt) {
      int ncol = (nbBase + nt) * 16 + (lane & 15);
      float bn = bias[ncol];
#pragma unroll
      for (int r = 0; r < 8; ++r) {
        int row = rowBase + wm * 64 + mt * 16 + r + hof;
        float v = acc[mt][nt][r] + bn;
        if (GELU) v = 0.5f * v * (1.0f + erff(v * 0.70710678118f));
        Y[(size_t)row * NOUT + ncol] = (_Float16)v;
      }
    }
  }
}

// =====================================================================
// FC2 GEMM: out_NCHW = x1 + (h1 @ fc2 + bias); LDS transpose for
// coalesced NHWC->NCHW final store.
// =====================================================================
union __align__(16) FC2Smem {
  _Float16 As[2][128][40];
  float Ct[128][129];
};

__global__ __launch_bounds__(256) void gemm_fc2(const _Float16* __restrict__ A,  // [M][1024]
                                                const _Float16* __restrict__ Bsw,
                                                const float* __restrict__ bias,  // [256]
                                                const float* __restrict__ x1,    // [M][256]
                                                float* __restrict__ out) {       // [B,C,H,W]
  constexpr int KDIM = 1024, KT = 32, NOUT = 256;
  __shared__ FC2Smem sm;
  const int tid = threadIdx.x;
  const int lane = tid & 31;
  const int wave = tid >> 5;
  const int wm = wave & 1;
  const int wn = wave >> 1;
  const int rowBase = blockIdx.x * 128;
  const int nbBase = blockIdx.y * 8 + wn * 2;
  v8f acc[4][2] = {};

  auto stage = [&](int buf, int kt) {
    const int k0 = kt * 32;
#pragma unroll
    for (int i = 0; i < 2; ++i) {
      int cid = tid + i * 256;
      int r = cid >> 2;
      int ch = (cid & 3) << 3;
      const _Float16* src = A + (size_t)(rowBase + r) * KDIM + k0 + ch;
      _Float16* dst = &sm.As[buf][r][ch];
#if ASYNC_COPY
      async_b128(src, dst);
#else
      *(half8*)dst = *(const half8*)src;
#endif
    }
  };

  stage(0, 0);
  for (int kt = 0; kt < KT; ++kt) {
    const int buf = kt & 1;
#if ASYNC_COPY
    async_wait0();
#endif
    __syncthreads();
    if (kt + 1 < KT) stage(buf ^ 1, kt + 1);

    const size_t bbase = ((size_t)kt * (NOUT >> 4) + nbBase) * 512 + lane * 16;
    v16h bf0 = *(const v16h*)(Bsw + bbase);
    v16h bf1 = *(const v16h*)(Bsw + bbase + 512);

    const int arow = wm * 64 + (lane & 15);
    const int hof = (lane >> 4) * 16;
    v16h af[4];
#pragma unroll
    for (int mt = 0; mt < 4; ++mt) {
      half8 a0 = *(const half8*)&sm.As[buf][arow + mt * 16][hof];
      half8 a1 = *(const half8*)&sm.As[buf][arow + mt * 16][hof + 8];
      af[mt] = __builtin_shufflevector(a0, a1, 0, 1, 2, 3, 4, 5, 6, 7,
                                       8, 9, 10, 11, 12, 13, 14, 15);
    }
#pragma unroll
    for (int mt = 0; mt < 4; ++mt) {
      acc[mt][0] = __builtin_amdgcn_wmma_f32_16x16x32_f16(false, af[mt], false, bf0,
                                                          (short)0, acc[mt][0], false, false);
      acc[mt][1] = __builtin_amdgcn_wmma_f32_16x16x32_f16(false, af[mt], false, bf1,
                                                          (short)0, acc[mt][1], false, false);
    }
    __syncthreads();
  }

  // bias + residual into LDS tile (fp32), then coalesced NCHW store
  const int hof = (lane >> 4) << 3;
#pragma unroll
  for (int mt = 0; mt < 4; ++mt) {
#pragma unroll
    for (int nt = 0; nt < 2; ++nt) {
      int nlocal = (wn * 2 + nt) * 16 + (lane & 15);
      int nglob = blockIdx.y * 128 + nlocal;
      float bn = bias[nglob];
#pragma unroll
      for (int r = 0; r < 8; ++r) {
        int mloc = wm * 64 + mt * 16 + r + hof;
        int row = rowBase + mloc;
        float v = acc[mt][nt][r] + bn + x1[(size_t)row * 256 + nglob];
        sm.Ct[mloc][nlocal] = v;
      }
    }
  }
  __syncthreads();
  const int tok = tid & 127;
  const int token = rowBase + tok;
  const int bimg = token >> 14;
  const int hw = token & 16383;
  for (int nl = (tid >> 7); nl < 128; nl += 2) {
    int c = blockIdx.y * 128 + nl;
    out[((size_t)(bimg * 256 + c) << 14) + hw] = sm.Ct[tok][nl];
  }
}

// =====================================================================
// Host launcher
// =====================================================================
extern "C" void kernel_launch(void* const* d_in, const int* in_sizes, int n_in,
                              void* d_out, int out_size, void* d_ws, size_t ws_size,
                              hipStream_t stream) {
  const float* x     = (const float*)d_in[0];
  const float* wq    = (const float*)d_in[1];
  const float* bq    = (const float*)d_in[2];
  const float* wk    = (const float*)d_in[3];
  const float* bk    = (const float*)d_in[4];
  const float* wv    = (const float*)d_in[5];
  const float* bv    = (const float*)d_in[6];
  const float* ln1_g = (const float*)d_in[7];
  const float* ln1_b = (const float*)d_in[8];
  const float* ln2_g = (const float*)d_in[9];
  const float* ln2_b = (const float*)d_in[10];
  const float* fc1_w = (const float*)d_in[11];
  const float* fc1_b = (const float*)d_in[12];
  const float* fc2_w = (const float*)d_in[13];
  const float* fc2_b = (const float*)d_in[14];
  float* out = (float*)d_out;

  char* ws = (char*)d_ws;
  size_t off = 0;
  auto alloc = [&](size_t bytes) { size_t o = off; off += (bytes + 255) & ~(size_t)255; return o; };

  _Float16* n1h  = (_Float16*)(ws + alloc((size_t)NTOK * 256 * 2));  // 32 MB (reused as n2)
  _Float16* qkvh = (_Float16*)(ws + alloc((size_t)NTOK * 768 * 2));  // 96 MB
  float*    xc   = (float*)   (ws + alloc((size_t)NTOK * 256 * 4));  // 64 MB
  float*    x1   = (float*)   (ws + alloc((size_t)NTOK * 256 * 4));  // 64 MB
  _Float16* wqkv = (_Float16*)(ws + alloc((size_t)768 * 256 * 2));
  _Float16* fc1s = (_Float16*)(ws + alloc((size_t)256 * 1024 * 2));
  _Float16* fc2s = (_Float16*)(ws + alloc((size_t)1024 * 256 * 2));
  float*    bqkv = (float*)   (ws + alloc((size_t)768 * 4));
  // Aliases (lifetimes disjoint): n2 over n1; mlp hidden over qkv + first 32MB of xc
  _Float16* n2h = n1h;
  _Float16* h1  = qkvh;   // needs 128 MB: qkv(96) + xc(32) both dead by then

  // 1. weight prep (fp32 -> f16 WMMA-B fragment layout; stays hot in L2)
  swizzle_qkv<<<dim3(768), 256, 0, stream>>>(wq, wk, wv, wqkv);
  swizzle_w  <<<dim3(1024), 256, 0, stream>>>(fc1_w, fc1s, 256, 1024);
  swizzle_w  <<<dim3(1024), 256, 0, stream>>>(fc2_w, fc2s, 1024, 256);
  bias_cat   <<<dim3(3), 256, 0, stream>>>(bq, bk, bv, bqkv);

  // 2. LN1 + transpose to channels-last
  ln1_transpose<<<dim3(1024), 256, 0, stream>>>(x, ln1_g, ln1_b, xc, n1h);

  // 3. fused QKV projection (WMMA)
  gemm_f16out<256, false><<<dim3(512, 6), 256, 0, stream>>>(n1h, wqkv, bqkv, qkvh, 768);

  // 4. 3x3 neighborhood attention + residual 1
  attn_kernel<<<dim3(8192), 256, 0, stream>>>(qkvh, xc, x1);

  // 5. LN2
  ln_rows<<<dim3(8192), 256, 0, stream>>>(x1, ln2_g, ln2_b, n2h);

  // 6. FC1 + exact GELU (WMMA)
  gemm_f16out<256, true><<<dim3(512, 8), 256, 0, stream>>>(n2h, fc1s, fc1_b, h1, 1024);

  // 7. FC2 + residual 2 + NCHW output (WMMA)
  gemm_fc2<<<dim3(512, 2), 256, 0, stream>>>(h1, fc2s, fc2_b, x1, out);

  (void)in_sizes; (void)n_in; (void)out_size; (void)ws_size;
}
